// ReferenceBRNN_84937273246211
// MI455X (gfx1250) — 8-bit matrix operand order repaired and checked by reading, compiled, not run on hardware
//
#include <hip/hip_runtime.h>
#include <hip/hip_bf16.h>


typedef int   v16i __attribute__((ext_vector_type(16)));
typedef float v8f  __attribute__((ext_vector_type(8)));

#define CARRY   384
#define READ    128
#define VOCAB   128
#define DMODEL  512
#define NFF     7
#define BATCH   256
#define TLEN    2048
#define MROWS   16
#define XSTRIDE 528
#define NTILES  32
#define KCH     4
#define LSTRIDE ((size_t)NTILES * KCH * 2048)

#define WPACK_BYTES (NFF * DMODEL * DMODEL)
#define HPACK_BYTES (READ * VOCAB)
#define EMB_BYTES   (VOCAB * READ)

#define FP8_P1 ((signed char)0x38)
#define FP8_M1 ((signed char)0xB8)

__device__ __forceinline__ signed char pm1_fp8(float f) {
    return (f >= 0.0f) ? FP8_P1 : FP8_M1;
}

__global__ void pack_ff_kernel(const float* __restrict__ ff,
                               unsigned int* __restrict__ wpack) {
    int w = blockIdx.x * blockDim.x + threadIdx.x;
    if (w >= NFF * NTILES * KCH * 32 * 16) return;
    int v    = w & 15;
    int lane = (w >> 4) & 31;
    int kc   = (w >> 9) & 3;
    int nt   = (w >> 11) & 31;
    int l    = w >> 16;
    int col  = nt * 16 + (lane & 15);
    int kb   = (v >> 1) * 16 + ((lane & 16) ? 8 : 0) + (v & 1) * 4;
    int K    = kc * 128 + kb;
    const float* base = ff + ((size_t)l * DMODEL + K) * DMODEL + col;
    unsigned int word = 0;
    #pragma unroll
    for (int j = 0; j < 4; ++j) {
        unsigned int b = (unsigned char)pm1_fp8(base[(size_t)j * DMODEL]);
        word |= b << (8 * j);
    }
    wpack[w] = word;
}

__global__ void pack_head_kernel(const float* __restrict__ head,
                                 unsigned int* __restrict__ hpack) {
    int w = blockIdx.x * blockDim.x + threadIdx.x;
    if (w >= 8 * 32 * 16) return;
    int v    = w & 15;
    int lane = (w >> 4) & 31;
    int nt   = w >> 9;
    int col  = nt * 16 + (lane & 15);
    int K    = (v >> 1) * 16 + ((lane & 16) ? 8 : 0) + (v & 1) * 4;
    unsigned int word = 0;
    #pragma unroll
    for (int j = 0; j < 4; ++j) {
        unsigned int b = (unsigned char)pm1_fp8(head[(size_t)(K + j) * VOCAB + col]);
        word |= b << (8 * j);
    }
    hpack[w] = word;
}

__global__ void pack_embed_kernel(const float* __restrict__ embed,
                                  unsigned int* __restrict__ emb) {
    int w = blockIdx.x * blockDim.x + threadIdx.x;
    if (w >= EMB_BYTES / 4) return;
    unsigned int word = 0;
    #pragma unroll
    for (int j = 0; j < 4; ++j) {
        unsigned int b = (unsigned char)pm1_fp8(embed[(size_t)w * 4 + j]);
        word |= b << (8 * j);
    }
    emb[w] = word;
}

__global__ __launch_bounds__(1024, 1)
void brnn_kernel(const int*         __restrict__ tokens,
                 const float*       __restrict__ initial,
                 const int*         __restrict__ thresh,
                 const signed char* __restrict__ wpack,
                 const signed char* __restrict__ hpack,
                 const signed char* __restrict__ emb,
                 float*             __restrict__ out)
{
    __shared__ __align__(16) signed char xs[2][MROWS][XSTRIDE];

    const int tid  = threadIdx.x;
    const int wave = tid >> 5;
    const int lane = tid & 31;
    const int b0   = blockIdx.x * MROWS;

    const int colInTile = lane & 15;
    const int half      = lane >> 4;
    const int myCol     = wave * 16 + colInTile;

    for (int u = tid; u < MROWS * (CARRY / 4); u += 1024) {
        int row = u / (CARRY / 4);
        int j   = u % (CARRY / 4);
        unsigned int word = 0;
        #pragma unroll
        for (int k = 0; k < 4; ++k)
            word |= ((unsigned int)(unsigned char)pm1_fp8(initial[j * 4 + k])) << (8 * k);
        *(unsigned int*)&xs[0][row][j * 4] = word;
    }

    v16i hB = {};
    if (wave < 8)
        hB = *(const v16i*)(hpack + (size_t)wave * 2048 + lane * 64);

    int cur = 0;
    #pragma unroll 1
    for (int t = 0; t < TLEN; ++t) {
        unsigned int lz = 0;
        asm volatile("" : "+s"(lz));
        const signed char* wrun = wpack + lz;

        if (tid < MROWS * (READ / 4)) {
            int row   = tid >> 5;
            int j     = tid & 31;
            int token = tokens[(size_t)(b0 + row) * TLEN + t];
            int wrd   = ((const int*)emb)[token * (READ / 4) + j];
            *(int*)&xs[cur][row][CARRY + j * 4] = wrd;
        }
        __syncthreads();

        #pragma unroll 1
        for (int l = 0; l < NFF; ++l) {
            const signed char* wl = wrun + ((size_t)l * NTILES + wave) * (KCH * 2048);

            if (l + 1 < NFF)
                __builtin_prefetch(wl + LSTRIDE + lane * 64, 0, 3);

            float thf = (float)thresh[l * DMODEL + myCol];

            v8f acc = {};
            #pragma unroll
            for (int kc = 0; kc < KCH; ++kc) {
                union { v16i v; long long q[8]; } A;
                const signed char* ar = &xs[cur][lane & 15][kc * 128 + half * 8];
                A.q[0] = *(const long long*)(ar + 0);
                A.q[1] = *(const long long*)(ar + 16);
                A.q[2] = *(const long long*)(ar + 32);
                A.q[3] = *(const long long*)(ar + 48);
                A.q[4] = *(const long long*)(ar + 64);
                A.q[5] = *(const long long*)(ar + 80);
                A.q[6] = *(const long long*)(ar + 96);
                A.q[7] = *(const long long*)(ar + 112);

                v16i B = *(const v16i*)(wl + (size_t)kc * 2048 + lane * 64);

                acc = __builtin_amdgcn_wmma_f32_16x16x128_fp8_fp8(
                          A.v, B, (short)0, acc, false, false);
            }

            int nxt = cur ^ 1;
            #pragma unroll
            for (int j = 0; j < 8; ++j) {
                int row = j + half * 8;
                xs[nxt][row][myCol] = (acc[j] >= thf) ? FP8_P1 : FP8_M1;
            }
            __syncthreads();
            cur = nxt;
        }

        if (wave < 8) {
            union { v16i v; long long q[8]; } A;
            const signed char* ar = &xs[cur][lane & 15][CARRY + half * 8];
            A.q[0] = *(const long long*)(ar + 0);
            A.q[1] = *(const long long*)(ar + 16);
            A.q[2] = *(const long long*)(ar + 32);
            A.q[3] = *(const long long*)(ar + 48);
            A.q[4] = *(const long long*)(ar + 64);
            A.q[5] = *(const long long*)(ar + 80);
            A.q[6] = *(const long long*)(ar + 96);
            A.q[7] = *(const long long*)(ar + 112);

            v8f acc = {};
            acc = __builtin_amdgcn_wmma_f32_16x16x128_fp8_fp8(
                      A.v, hB, (short)0, acc, false, false);

            #pragma unroll
            for (int j = 0; j < 8; ++j) {
                int row = j + half * 8;
                out[((size_t)(b0 + row) * TLEN + t) * VOCAB + myCol] = acc[j];
            }
        }
        __syncthreads();
    }
}

extern "C" void kernel_launch(void* const* d_in, const int* in_sizes, int n_in,
                              void* d_out, int out_size, void* d_ws, size_t ws_size,
                              hipStream_t stream) {
    (void)in_sizes; (void)n_in; (void)out_size; (void)ws_size;

    const int*   tokens  = (const int*)d_in[0];
    const float* initial = (const float*)d_in[1];
    const float* embed   = (const float*)d_in[2];
    const float* ff      = (const float*)d_in[3];
    const float* head    = (const float*)d_in[4];
    const int*   thresh  = (const int*)d_in[5];
    float*       out     = (float*)d_out;

    signed char* wpack = (signed char*)d_ws;
    signed char* hpack = wpack + WPACK_BYTES;
    signed char* emb   = hpack + HPACK_BYTES;

    {
        int wWords = WPACK_BYTES / 4;
        pack_ff_kernel<<<(wWords + 255) / 256, 256, 0, stream>>>(
            ff, (unsigned int*)wpack);
        pack_head_kernel<<<(HPACK_BYTES / 4 + 255) / 256, 256, 0, stream>>>(
            head, (unsigned int*)hpack);
        pack_embed_kernel<<<(EMB_BYTES / 4 + 255) / 256, 256, 0, stream>>>(
            embed, (unsigned int*)emb);
    }

    brnn_kernel<<<BATCH / MROWS, 1024, 0, stream>>>(
        tokens, initial, thresh, wpack, hpack, emb, out);
}
